// GCNEncoder_6433861009969
// MI455X (gfx1250) — matrix operand order repaired and checked by reading, not compiled, not run on hardware
//
#include <hip/hip_runtime.h>
#include <hip/hip_bf16.h>

typedef __attribute__((ext_vector_type(16))) __bf16 v16bf;
typedef __attribute__((ext_vector_type(8)))  float  v8f;

#define NODES  131072
#define BGR    64
#define NPGR   2048
#define NEDGE  1048576
#define HIDD   128

// ---- CDNA5 async global->LDS copy (CDNA5 ISA) -------
__device__ __forceinline__ void async_copy_b128(void* lds_ptr, const void* gptr)
{
    // generic shared pointer: low 32 bits = LDS byte offset (aperture in high bits)
    unsigned lds_off = (unsigned)(uintptr_t)lds_ptr;
    asm volatile("global_load_async_to_lds_b128 %0, %1, off"
                 :: "v"(lds_off), "v"(gptr) : "memory");
}
__device__ __forceinline__ void wait_async_zero()
{
    asm volatile("s_wait_asynccnt 0x0" ::: "memory");
}

// ---- ISA fragment index mapping (CDNA5 ISA §7.12.2) ------------
// 16-bit A 16x32: lane m = lane&15, half = lane>>4:
//   a[0..7]  = A[m, k0 + half*8 .. +7],  a[8..15] = A[m, k0+16+half*8 .. +7]
// 16-bit B 32x16: lane n = lane&15: b[i] pairs with a[i]; this code uses K = k0 + half*16 + i for A and B
//   => with B stored transposed [N,K], fragment is 16 contiguous bf16.

__device__ __forceinline__ v16bf load_a_frag(const float* __restrict__ arow, int k0)
{
    const float4 f0 = *(const float4*)(arow + k0);
    const float4 f1 = *(const float4*)(arow + k0 + 4);
    const float4 f2 = *(const float4*)(arow + k0 + 8);
    const float4 f3 = *(const float4*)(arow + k0 + 12);
    v16bf a;
    a[0]  = (__bf16)f0.x; a[1]  = (__bf16)f0.y; a[2]  = (__bf16)f0.z; a[3]  = (__bf16)f0.w;
    a[4]  = (__bf16)f1.x; a[5]  = (__bf16)f1.y; a[6]  = (__bf16)f1.z; a[7]  = (__bf16)f1.w;
    a[8]  = (__bf16)f2.x; a[9]  = (__bf16)f2.y; a[10] = (__bf16)f2.z; a[11] = (__bf16)f2.w;
    a[12] = (__bf16)f3.x; a[13] = (__bf16)f3.y; a[14] = (__bf16)f3.z; a[15] = (__bf16)f3.w;
    return a;
}

// ---------------------------------------------------------------------------
// Block: 256 threads = 8 waves; block tile 128 rows x 64 cols.
// B slice (64 x K bf16) staged in LDS via async copy; A streamed from global
// with a one-step software pipeline on the A fragment.
template<int K>
__global__ void wmma_gemm_bf16(const float* __restrict__ A, int lda,
                               const __bf16* __restrict__ Bt,   // [N,K] bf16
                               const float* __restrict__ bias,
                               float* __restrict__ C, int ldc,
                               int M, int N, int relu)
{
    __shared__ __align__(16) __bf16 sB[64 * K];

    const int nb = N >> 6;                       // 64-col tiles across N
    const int bm = blockIdx.x / nb;
    const int bn = blockIdx.x % nb;

    // ---- stage B tile: 64*K bf16, 16B chunks, async to LDS
    {
        const char* bsrc = (const char*)(Bt + (long)bn * 64 * K);
        char*       bdst = (char*)sB;
        constexpr int CHUNKS = (64 * K * 2) / 16;
        constexpr int ITER   = CHUNKS / 256;
#pragma unroll
        for (int i = 0; i < ITER; ++i) {
            int c = threadIdx.x + (i << 8);
            async_copy_b128(bdst + c * 16, bsrc + c * 16);
        }
        wait_async_zero();
    }
    __syncthreads();

    const int wv = threadIdx.x >> 5;
    const int m0 = bm * 128 + wv * 16;
    if (m0 < M) {                                // wave-uniform: EXEC stays all-1s
        const int lane = threadIdx.x & 31;
        const int half = lane >> 4;
        const int mn   = lane & 15;

        const float*  arow = A  + (long)(m0 + mn) * lda + (half << 4);
        const __bf16* brow = sB + (long)mn * K + (half << 4);

        v8f acc[4] = {{}, {}, {}, {}};
        v16bf a_cur = load_a_frag(arow, 0);
#pragma unroll
        for (int k0 = 0; k0 < K; k0 += 32) {
            // batch all four B fragments: one dscnt wait covers four WMMAs
            v16bf b0 = *(const v16bf*)(brow + (0 * 16) * K + k0);
            v16bf b1 = *(const v16bf*)(brow + (1 * 16) * K + k0);
            v16bf b2 = *(const v16bf*)(brow + (2 * 16) * K + k0);
            v16bf b3 = *(const v16bf*)(brow + (3 * 16) * K + k0);
            // prefetch next A fragment into fresh registers (kills WAR v_nops)
            v16bf a_use = a_cur;
            if (k0 + 32 < K) a_cur = load_a_frag(arow, k0 + 32);
            acc[0] = __builtin_amdgcn_wmma_f32_16x16x32_bf16(
                         false, a_use, false, b0, (short)0, acc[0], false, false);
            acc[1] = __builtin_amdgcn_wmma_f32_16x16x32_bf16(
                         false, a_use, false, b1, (short)0, acc[1], false, false);
            acc[2] = __builtin_amdgcn_wmma_f32_16x16x32_bf16(
                         false, a_use, false, b2, (short)0, acc[2], false, false);
            acc[3] = __builtin_amdgcn_wmma_f32_16x16x32_bf16(
                         false, a_use, false, b3, (short)0, acc[3], false, false);
        }
#pragma unroll
        for (int nt = 0; nt < 4; ++nt) {
            const int n = bn * 64 + (nt << 4) + mn;
            const float bv = bias ? bias[n] : 0.0f;
#pragma unroll
            for (int r = 0; r < 8; ++r) {
                const int m = m0 + r + (half << 3);
                float v = acc[nt][r] + bv;
                if (relu) v = fmaxf(v, 0.0f);
                C[(long)m * ldc + n] = v;
            }
        }
    }
}

// ---------------------------------------------------------------------------
__global__ void cvt_w_transpose(const float* __restrict__ w, __bf16* __restrict__ wt,
                                int K, int N)
{
    int i = blockIdx.x * blockDim.x + threadIdx.x;
    if (i >= K * N) return;
    int k = i / N, n = i - k * N;
    wt[(long)n * K + k] = (__bf16)w[i];
}

__global__ void deg_init(float* __restrict__ deg, int n)
{
    int i = blockIdx.x * blockDim.x + threadIdx.x;
    if (i < n) deg[i] = 1.0f;                 // self-loop
}

__global__ void deg_count(const int* __restrict__ src, const int* __restrict__ dst,
                          float* __restrict__ deg1, float* __restrict__ deg2)
{
    int e = blockIdx.x * blockDim.x + threadIdx.x;   // exact NEDGE
    atomicAdd(&deg1[dst[e]], 1.0f);
    atomicAdd(&deg2[src[e]], 1.0f);
}

__global__ void deg_rsqrt(float* __restrict__ d, int n)
{
    int i = blockIdx.x * blockDim.x + threadIdx.x;
    if (i < n) d[i] = rsqrtf(d[i]);
}

__global__ void embed_gather(const float* __restrict__ emb, const int* __restrict__ tok,
                             float* __restrict__ x)
{
    int t = blockIdx.x * blockDim.x + threadIdx.x;   // NODES*32 exact
    int node = t >> 5, c4 = (t & 31) << 2;
    const float4 v = *(const float4*)(emb + (long)tok[node] * HIDD + c4);
    *(float4*)(x + (long)node * HIDD + c4) = v;
}

// out[n,c] = h[n,c]*dis[n]^2 + bias[c]   (self-loop term + bias init)
__global__ void conv_self(const float* __restrict__ h, int hstride,
                          const float* __restrict__ dis, const float* __restrict__ bias,
                          float* __restrict__ out, int ostride)
{
    int idx = blockIdx.x * blockDim.x + threadIdx.x;  // NODES*128 exact
    int node = idx >> 7, c = idx & 127;
    float w = dis[node];
    out[(long)node * ostride + c] = h[(long)node * hstride + c] * w * w + bias[c];
}

// out[d] += h[s] * dis[s]*dis[d]  over edges; 32 lanes x float4 per edge
__global__ void conv_edge(const float* __restrict__ h, int hstride,
                          const int* __restrict__ s, const int* __restrict__ d,
                          const float* __restrict__ dis,
                          float* __restrict__ out, int ostride)
{
    int t = blockIdx.x * blockDim.x + threadIdx.x;    // NEDGE*32 exact
    int e = t >> 5, c4 = (t & 31) << 2;
    int ss = s[e], dd = d[e];
    float nrm = dis[ss] * dis[dd];
    const float4 hv = *(const float4*)(h + (long)ss * hstride + c4);
    float* o = out + (long)dd * ostride + c4;
    atomicAdd(o + 0, hv.x * nrm);
    atomicAdd(o + 1, hv.y * nrm);
    atomicAdd(o + 2, hv.z * nrm);
    atomicAdd(o + 3, hv.w * nrm);
}

__global__ void pool_mean(const float* __restrict__ dense, float* __restrict__ gemb)
{
    int b = blockIdx.x, c = threadIdx.x;              // 64 blocks x 256 threads
    const float* p = dense + (long)b * NPGR * 256 + c;
    float s = 0.0f;
    for (int i = 0; i < NPGR; ++i) s += p[(long)i * 256];
    gemb[b * 256 + c] = s * (1.0f / NPGR);
}

__global__ void mask_fill(float* __restrict__ m, int n)
{
    int i = blockIdx.x * blockDim.x + threadIdx.x;
    if (i < n) m[i] = 1.0f;
}

// ---------------------------------------------------------------------------
extern "C" void kernel_launch(void* const* d_in, const int* in_sizes, int n_in,
                              void* d_out, int out_size, void* d_ws, size_t ws_size,
                              hipStream_t stream)
{
    // --- inputs (setup_inputs dict order, params flattened in declaration order)
    const float* emb    = (const float*)d_in[0];
    const float* g1W0   = (const float*)d_in[1];
    const float* g1b0   = (const float*)d_in[2];
    const float* g1W1   = (const float*)d_in[3];
    const float* g1b1   = (const float*)d_in[4];
    const float* g2W0   = (const float*)d_in[5];
    const float* g2b0   = (const float*)d_in[6];
    const float* g2W1   = (const float*)d_in[7];
    const float* g2b1   = (const float*)d_in[8];
    const float* fcW    = (const float*)d_in[9];
    const float* fcb    = (const float*)d_in[10];
    const float* wW[4]  = { (const float*)d_in[11], (const float*)d_in[13],
                            (const float*)d_in[15], (const float*)d_in[17] };
    const float* wb[4]  = { (const float*)d_in[12], (const float*)d_in[14],
                            (const float*)d_in[16], (const float*)d_in[18] };
    const int* tokens   = (const int*)d_in[19];
    const int* src      = (const int*)d_in[20];
    const int* dst      = src + NEDGE;

    // --- output layout (floats): dense_x | encoder_feature | mask | h1 h2 h3 h4
    float* out    = (float*)d_out;
    float* dense  = out;                          // [N,256] == x12 (x1/x2 home)
    float* enc    = out + (long)NODES * 256;      // [N,256]
    float* maskp  = enc + (long)NODES * 256;      // [B,NPG]
    float* h1     = maskp + (long)BGR * NPGR;     // [64,128] x4
    float* h2     = h1 + BGR * HIDD;
    float* h3     = h2 + BGR * HIDD;
    float* h4     = h3 + BGR * HIDD;

    // scratch inside enc region until fc GEMM overwrites it
    float* x_emb  = enc;                          // [N,128]
    float* h_tmp  = enc + (long)NODES * HIDD;     // [N,128]
    float* cur1   = dense;                        // stride 256, cols [0,128)
    float* cur2   = dense + HIDD;                 // stride 256, cols [128,256)

    // --- workspace: dis1 | dis2 | gemb | bf16 weights (transposed [N,K])
    float*  dis1  = (float*)d_ws;                 // NODES
    float*  dis2  = dis1 + NODES;                 // NODES
    float*  gemb  = dis2 + NODES;                 // 64*256
    __bf16* wt    = (__bf16*)(gemb + BGR * 256);
    __bf16* g1T0  = wt;                 // 128*128
    __bf16* g1T1  = g1T0 + 16384;
    __bf16* g2T0  = g1T1 + 16384;
    __bf16* g2T1  = g2T0 + 16384;
    __bf16* fcT   = g2T1 + 16384;       // 256*256
    __bf16* wT0   = fcT  + 65536;       // 256*128 x4
    __bf16* wT1   = wT0  + 32768;
    __bf16* wT2   = wT1  + 32768;
    __bf16* wT3   = wT2  + 32768;

    const int TB = 256;

    // 1) weight conversion (f32 [K,N] -> bf16 [N,K])
    cvt_w_transpose<<<(16384 + TB - 1) / TB, TB, 0, stream>>>(g1W0, g1T0, 128, 128);
    cvt_w_transpose<<<(16384 + TB - 1) / TB, TB, 0, stream>>>(g1W1, g1T1, 128, 128);
    cvt_w_transpose<<<(16384 + TB - 1) / TB, TB, 0, stream>>>(g2W0, g2T0, 128, 128);
    cvt_w_transpose<<<(16384 + TB - 1) / TB, TB, 0, stream>>>(g2W1, g2T1, 128, 128);
    cvt_w_transpose<<<(65536 + TB - 1) / TB, TB, 0, stream>>>(fcW,  fcT,  256, 256);
    cvt_w_transpose<<<(32768 + TB - 1) / TB, TB, 0, stream>>>(wW[0], wT0, 256, 128);
    cvt_w_transpose<<<(32768 + TB - 1) / TB, TB, 0, stream>>>(wW[1], wT1, 256, 128);
    cvt_w_transpose<<<(32768 + TB - 1) / TB, TB, 0, stream>>>(wW[2], wT2, 256, 128);
    cvt_w_transpose<<<(32768 + TB - 1) / TB, TB, 0, stream>>>(wW[3], wT3, 256, 128);

    // 2) degrees -> rsqrt
    deg_init<<<(2 * NODES) / TB, TB, 0, stream>>>(dis1, 2 * NODES);
    deg_count<<<NEDGE / TB, TB, 0, stream>>>(src, dst, dis1, dis2);
    deg_rsqrt<<<(2 * NODES) / TB, TB, 0, stream>>>(dis1, 2 * NODES);

    // 3) embedding gather
    embed_gather<<<(NODES * 32) / TB, TB, 0, stream>>>(emb, tokens, x_emb);

    // GEMM grids: block tile = 128 rows x 64 cols
    const int gemmN128 = (NODES / 128) * (128 / 64);     // 2048 blocks
    const int gemmN256 = (NODES / 128) * (256 / 64);     // 4096 blocks
    const int gemmHead = 1 * (128 / 64);                 // 2 blocks (M=64)
    const int edgeBlk  = (NEDGE * 32) / TB;
    const int selfBlk  = (NODES * 128) / TB;

    // 4) stack 1 (messages src->dst, dis1)
    wmma_gemm_bf16<128><<<gemmN128, TB, 0, stream>>>(x_emb, 128, g1T0, nullptr,
                                                     h_tmp, 128, NODES, 128, 0);
    conv_self<<<selfBlk, TB, 0, stream>>>(h_tmp, 128, dis1, g1b0, cur1, 256);
    conv_edge<<<edgeBlk, TB, 0, stream>>>(h_tmp, 128, src, dst, dis1, cur1, 256);

    wmma_gemm_bf16<128><<<gemmN128, TB, 0, stream>>>(cur1, 256, g1T1, nullptr,
                                                     h_tmp, 128, NODES, 128, 0);
    conv_self<<<selfBlk, TB, 0, stream>>>(h_tmp, 128, dis1, g1b1, cur1, 256);
    conv_edge<<<edgeBlk, TB, 0, stream>>>(h_tmp, 128, src, dst, dis1, cur1, 256);

    // 5) stack 2 (messages dst->src, dis2)
    wmma_gemm_bf16<128><<<gemmN128, TB, 0, stream>>>(x_emb, 128, g2T0, nullptr,
                                                     h_tmp, 128, NODES, 128, 0);
    conv_self<<<selfBlk, TB, 0, stream>>>(h_tmp, 128, dis2, g2b0, cur2, 256);
    conv_edge<<<edgeBlk, TB, 0, stream>>>(h_tmp, 128, dst, src, dis2, cur2, 256);

    wmma_gemm_bf16<128><<<gemmN128, TB, 0, stream>>>(cur2, 256, g2T1, nullptr,
                                                     h_tmp, 128, NODES, 128, 0);
    conv_self<<<selfBlk, TB, 0, stream>>>(h_tmp, 128, dis2, g2b1, cur2, 256);
    conv_edge<<<edgeBlk, TB, 0, stream>>>(h_tmp, 128, dst, src, dis2, cur2, 256);

    // 6) encoder_feature = relu(dense @ fc + fcb)  (overwrites scratch region)
    wmma_gemm_bf16<256><<<gemmN256, TB, 0, stream>>>(dense, 256, fcT, fcb,
                                                     enc, 256, NODES, 256, 1);

    // 7) graph mean pool + 4 head GEMMs -> h1..h4
    pool_mean<<<BGR, 256, 0, stream>>>(dense, gemb);
    wmma_gemm_bf16<256><<<gemmHead, TB, 0, stream>>>(gemb, 256, wT0, wb[0], h1, 128, BGR, 128, 0);
    wmma_gemm_bf16<256><<<gemmHead, TB, 0, stream>>>(gemb, 256, wT1, wb[1], h2, 128, BGR, 128, 0);
    wmma_gemm_bf16<256><<<gemmHead, TB, 0, stream>>>(gemb, 256, wT2, wb[2], h3, 128, BGR, 128, 0);
    wmma_gemm_bf16<256><<<gemmHead, TB, 0, stream>>>(gemb, 256, wT3, wb[3], h4, 128, BGR, 128, 0);

    // 8) valid_mask = all true
    mask_fill<<<(BGR * NPGR) / TB, TB, 0, stream>>>(maskp, BGR * NPGR);
}
